// SphericalFNO_53815940218924
// MI455X (gfx1250) — matrix operand order repaired and checked by reading, not compiled, not run on hardware
//
#include <hip/hip_runtime.h>
#include <stdint.h>

// ---------------- problem constants ----------------
#define NLAT   96
#define NLON   192
#define NPIX   (NLAT*NLON)     // 18432
#define MODES  48
#define KTOT   2304            // (LMAX+1)^2
#define HID    128
#define INC    8
#define OUTC   8
#define BATCH  16
#define NLAYERS 4
#define BH     (BATCH*HID)     // 2048

// ---------------- CDNA5 WMMA types ----------------
typedef __attribute__((ext_vector_type(16))) __bf16 v16bf;
typedef __attribute__((ext_vector_type(8)))  float  v8f;

struct alignas(16) U128 { unsigned int a, b, c, d; };
struct alignas(32) FragBF { U128 lo, hi; };

__device__ __forceinline__ v16bf frag_cast(const FragBF& f) {
    return __builtin_bit_cast(v16bf, f);
}

// round-to-nearest-even f32 -> bf16 (bit trick, toolchain-agnostic)
__device__ __forceinline__ unsigned short f2bf(float f) {
    union { float f; uint32_t u; } v; v.f = f;
    uint32_t r = v.u + 0x7fffu + ((v.u >> 16) & 1u);
    return (unsigned short)(r >> 16);
}

__device__ __forceinline__ float gelu_exact(float v) {
    return 0.5f * v * (1.0f + erff(v * 0.7071067811865476f));
}

// ---------------- gfx1250 async global->LDS path ----------------
#if __has_builtin(__builtin_amdgcn_global_load_async_to_lds_b128) && \
    __has_builtin(__builtin_amdgcn_s_wait_asynccnt)
#define USE_ASYNC_LDS 1
#else
#define USE_ASYNC_LDS 0
#endif

#define AS1 __attribute__((address_space(1)))
#define AS3 __attribute__((address_space(3)))

typedef int vi4 __attribute__((vector_size(16)));   // b128 payload type

#if USE_ASYNC_LDS
// Copy 32 contiguous bytes global->LDS asynchronously (2 x b128 ops).
// INST_OFFSET is added to BOTH the global and LDS addresses (ISA ch.8),
// so a single base pair + imm 16 covers the second beat.
__device__ __forceinline__ void async_cp32(const void* g, void* l) {
    AS1 vi4* gp = (AS1 vi4*)(uintptr_t)g;
    AS3 vi4* lp = (AS3 vi4*)(unsigned int)(uintptr_t)l;
    __builtin_amdgcn_global_load_async_to_lds_b128(gp, lp, 0, 0);
    __builtin_amdgcn_global_load_async_to_lds_b128(gp, lp, 16, 0);
}
#endif

// ============================================================
// 1) Y conversion: Ybf[k<48][ij] (fwd B-operand) and full
//    transpose Ytbf[ij][k] (inverse B-operand), both bf16.
// ============================================================
__global__ void __launch_bounds__(256)
k_yconv(const float* __restrict__ Y,
        unsigned short* __restrict__ Ybf,     // [48][NPIX]
        unsigned short* __restrict__ Ytbf) {  // [NPIX][KTOT]
    __shared__ float tile[32][33];
    const int pb = blockIdx.x * 32;           // pixel tile (576)
    const int kb = blockIdx.y * 32;           // mode tile  (72)
    const int t  = threadIdx.x;
    const int r  = t >> 3;                    // 0..31
    const int c4 = (t & 7) << 2;              // 0,4,..,28
    const float* src = Y + (size_t)(kb + r) * NPIX + pb + c4;
    float v0 = src[0], v1 = src[1], v2 = src[2], v3 = src[3];
    tile[r][c4+0] = v0; tile[r][c4+1] = v1;
    tile[r][c4+2] = v2; tile[r][c4+3] = v3;
    if (kb + r < MODES) {
        unsigned short* d = Ybf + (size_t)(kb + r) * NPIX + pb + c4;
        d[0] = f2bf(v0); d[1] = f2bf(v1); d[2] = f2bf(v2); d[3] = f2bf(v3);
    }
    __syncthreads();
    unsigned short* d = Ytbf + (size_t)(pb + r) * KTOT + kb + c4;
    d[0] = f2bf(tile[c4+0][r]); d[1] = f2bf(tile[c4+1][r]);
    d[2] = f2bf(tile[c4+2][r]); d[3] = f2bf(tile[c4+3][r]);
}

// ============================================================
// 2) Encoder: hq[b*HID+h][ij] = bf16( qw[lat] * (in_w·x + in_b) )
// ============================================================
__global__ void __launch_bounds__(256)
k_encode(const float* __restrict__ x, const float* __restrict__ in_w,
         const float* __restrict__ in_b, const float* __restrict__ qw,
         unsigned short* __restrict__ hq) {
    __shared__ float sx[INC][128];
    __shared__ float sw[HID * INC];
    __shared__ float sb[HID];
    __shared__ float sqw[128];
    const int b   = blockIdx.x / (NPIX / 128);
    const int ij0 = (blockIdx.x % (NPIX / 128)) * 128;
    const int t   = threadIdx.x;
    for (int i = t; i < HID * INC; i += 256) sw[i] = in_w[i];
    if (t < HID) sb[t] = in_b[t];
    for (int i = t; i < INC * 128; i += 256) {
        int c = i >> 7, jj = i & 127;
        sx[c][jj] = x[((size_t)(b * INC + c)) * NPIX + ij0 + jj];
    }
    if (t < 128) sqw[t] = qw[(ij0 + t) / NLON];
    __syncthreads();
    for (int u = 0; u < 64; ++u) {
        int idx = t + u * 256;         // 0..16383 = 128h x 128ij
        int h = idx >> 7, jj = idx & 127;
        float s = sb[h];
#pragma unroll
        for (int c = 0; c < INC; ++c) s += sw[h * INC + c] * sx[c][jj];
        hq[(size_t)(b * HID + h) * NPIX + ij0 + jj] = f2bf(s * sqw[jj]);
    }
}

// ============================================================
// 3) Forward SHT (N=48 modes only), split-K with f32 atomics.
//    C[bh][n] += sum_k hq[bh][k] * Ybf[n][k]
//    128 threads = 4 waves, BM=64, BN=48, K-slab = 768.
// ============================================================
__global__ void __launch_bounds__(128)
k_fwd_sht(const unsigned short* __restrict__ hq,
          const unsigned short* __restrict__ Ybf,
          float* __restrict__ cr) {
    __shared__ alignas(16) unsigned short As[64 * 32];
    __shared__ alignas(16) unsigned short Bs[48 * 32];
    const int t = threadIdx.x;
    const int wave = t >> 5, lane = t & 31;
    const int lmod = lane & 15, lhalf = lane >> 4;
    const int m0 = blockIdx.x * 64;
    const int k0 = blockIdx.y * 768;
    const int lrow = t >> 1, lcol = (t & 1) * 16;
    v8f acc[3];
#pragma unroll
    for (int f = 0; f < 3; ++f) acc[f] = (v8f){0,0,0,0,0,0,0,0};

    for (int ks = 0; ks < 24; ++ks) {
        const int kk = k0 + ks * 32;
        const unsigned short* gA = hq + (size_t)(m0 + lrow) * NPIX + kk + lcol;
        const unsigned short* gB = Ybf + (size_t)lrow * NPIX + kk + lcol;
#if USE_ASYNC_LDS
        async_cp32(gA, As + lrow * 32 + lcol);
        if (t < 96) async_cp32(gB, Bs + lrow * 32 + lcol);
        __builtin_amdgcn_s_wait_asynccnt(0);
#else
        {   const U128* g = (const U128*)gA;
            U128* s = (U128*)(As + lrow * 32 + lcol);
            s[0] = g[0]; s[1] = g[1]; }
        if (t < 96) {
            const U128* g = (const U128*)gB;
            U128* s = (U128*)(Bs + lrow * 32 + lcol);
            s[0] = g[0]; s[1] = g[1]; }
#endif
        if (ks + 1 < 24) {
            __builtin_prefetch(hq + (size_t)(m0 + lrow) * NPIX + kk + 32 + lcol, 0, 0);
        }
        __syncthreads();
        // A fragment: wave owns rows [wave*16, +16)
        FragBF af;
        const unsigned short* abase = As + (wave * 16 + lmod) * 32 + lhalf * 8;
        af.lo = *(const U128*)(abase);
        af.hi = *(const U128*)(abase + 16);
#pragma unroll
        for (int f = 0; f < 3; ++f) {
            FragBF bf_;
            const unsigned short* bbase = Bs + (f * 16 + lmod) * 32 + lhalf * 8;
            bf_.lo = *(const U128*)(bbase);
            bf_.hi = *(const U128*)(bbase + 16);
            acc[f] = __builtin_amdgcn_wmma_f32_16x16x32_bf16(
                false, frag_cast(af), false, frag_cast(bf_),
                (short)0, acc[f], false, false);
        }
        __syncthreads();
    }
#pragma unroll
    for (int f = 0; f < 3; ++f) {
        const int col = f * 16 + lmod;
#pragma unroll
        for (int r = 0; r < 8; ++r) {
            const int row = m0 + wave * 16 + r + lhalf * 8;
            atomicAdd(&cr[(size_t)row * KTOT + col], acc[f][r]);
        }
    }
}

// ============================================================
// 4) One FNO layer on the live 48 modes (in-place, per-mode block)
// ============================================================
__global__ void __launch_bounds__(128)
k_layer(const float* __restrict__ wr, const float* __restrict__ wi,
        const float* __restrict__ sp_bias, const float* __restrict__ lin_w,
        const float* __restrict__ lin_b, float* __restrict__ cr,
        float* __restrict__ ci, int l) {
    __shared__ float xr[BATCH][HID], xi[BATCH][HID];
    __shared__ float yr[BATCH][HID], yi[BATCH][HID];
    const int m = blockIdx.x;      // mode
    const int t = threadIdx.x;     // channel j / o
    for (int b = 0; b < BATCH; ++b) {
        xr[b][t] = cr[(size_t)(b * HID + t) * KTOT + m];
        xi[b][t] = ci[(size_t)(b * HID + t) * MODES + m];
    }
    __syncthreads();
    float ar[BATCH], ai[BATCH];
#pragma unroll
    for (int b = 0; b < BATCH; ++b) { ar[b] = 0.f; ai[b] = 0.f; }
    for (int i = 0; i < HID; ++i) {
        const size_t widx = (((size_t)l * HID + i) * HID + t) * MODES + m;
        const float wrv = wr[widx], wiv = wi[widx];
#pragma unroll
        for (int b = 0; b < BATCH; ++b) {
            const float xrv = xr[b][i], xiv = xi[b][i];
            ar[b] += xrv * wrv - xiv * wiv;
            ai[b] += xrv * wiv + xiv * wrv;
        }
    }
    const float sbv = sp_bias[((size_t)l * HID + t) * KTOT + m];
    for (int b = 0; b < BATCH; ++b) { yr[b][t] = ar[b] + sbv; yi[b][t] = ai[b]; }
    __syncthreads();
    float zr[BATCH], zi[BATCH];
#pragma unroll
    for (int b = 0; b < BATCH; ++b) { zr[b] = 0.f; zi[b] = 0.f; }
    for (int i = 0; i < HID; ++i) {
        const float lw = lin_w[((size_t)l * HID + t) * HID + i];
#pragma unroll
        for (int b = 0; b < BATCH; ++b) {
            zr[b] += lw * yr[b][i];
            zi[b] += lw * yi[b][i];
        }
    }
    const float lb = lin_b[l * HID + t];
    for (int b = 0; b < BATCH; ++b) {
        cr[(size_t)(b * HID + t) * KTOT + m] = gelu_exact(zr[b] + lb);
        ci[(size_t)(b * HID + t) * MODES + m] = gelu_exact(zi[b]);
    }
}

// ============================================================
// 5) Final-layer spectral tail (k>=48): batch-independent,
//    broadcast gelu(lin_w[3]·sp_bias[3] + lin_b[3]) to all b.
// ============================================================
__global__ void __launch_bounds__(128)
k_tail(const float* __restrict__ sp_bias, const float* __restrict__ lin_w,
       const float* __restrict__ lin_b, float* __restrict__ cr) {
    __shared__ float col[HID];
    const int k = MODES + blockIdx.x;      // 48..2303
    const int t = threadIdx.x;             // o
    const int l = NLAYERS - 1;
    col[t] = sp_bias[((size_t)l * HID + t) * KTOT + k];
    __syncthreads();
    float s = lin_b[l * HID + t];
    for (int i = 0; i < HID; ++i)
        s += lin_w[((size_t)l * HID + t) * HID + i] * col[i];
    const float g = gelu_exact(s);
    for (int b = 0; b < BATCH; ++b)
        cr[(size_t)(b * HID + t) * KTOT + k] = g;
}

// ============================================================
// 6) cr -> bf16
// ============================================================
__global__ void k_crbf(const float* __restrict__ cr,
                       unsigned short* __restrict__ crbf) {
    const size_t i = (size_t)blockIdx.x * 256 + threadIdx.x;
    crbf[i] = f2bf(cr[i]);
}

// ============================================================
// 7) Inverse SHT fused with output projection.
//    grid[bh][ij] = sum_k crbf[bh][k] * Ytbf[ij][k]   (WMMA bf16)
//    out[b][o][ij] = sum_h out_w[o][h]*grid + out_b[o] (LDS epilogue)
//    256 threads = 8 waves (4 M x 2 N), BM=128 (= one batch), BN=128.
//    Async double-buffered LDS staging (ASYNCcnt) when available.
// ============================================================
#define INV_KSTEPS (KTOT / 32)   // 72
// smem layout (aliased): [0,32KB) double-buffered A/B tiles,
// [0,34816) Cls (reuses tile space after K loop), [34816,+4KB) Wout.
#define SMEM_CLS    34816
#define SMEM_TOTAL  (SMEM_CLS + OUTC * HID * 4)

__global__ void __launch_bounds__(256)
k_inv_proj(const unsigned short* __restrict__ crbf,
           const unsigned short* __restrict__ Ytbf,
           const float* __restrict__ out_w, const float* __restrict__ out_b,
           float* __restrict__ out) {
    __shared__ alignas(16) char smem[SMEM_TOTAL];
    unsigned short* AsBuf[2] = { (unsigned short*)(smem + 0),
                                 (unsigned short*)(smem + 8192) };
    unsigned short* BsBuf[2] = { (unsigned short*)(smem + 16384),
                                 (unsigned short*)(smem + 24576) };
    float* Cls  = (float*)smem;                 // 128 x 68 (aliases tiles)
    float* Wout = (float*)(smem + SMEM_CLS);    // 8 x 128

    const int t = threadIdx.x;
    const int wave = t >> 5, lane = t & 31;
    const int wm = wave & 3, wn = wave >> 2;
    const int lmod = lane & 15, lhalf = lane >> 4;
    const int b  = blockIdx.y;            // batch = M block
    const int n0 = blockIdx.x * 128;      // pixel block
    const int lrow = t >> 1, lcol = (t & 1) * 16;

    for (int i = t; i < OUTC * HID; i += 256) Wout[i] = out_w[i];

    v8f acc[2][4];
#pragma unroll
    for (int i = 0; i < 2; ++i)
#pragma unroll
        for (int j = 0; j < 4; ++j) acc[i][j] = (v8f){0,0,0,0,0,0,0,0};

    const unsigned short* gA0 = crbf + (size_t)(b * HID + lrow) * KTOT + lcol;
    const unsigned short* gB0 = Ytbf + (size_t)(n0 + lrow) * KTOT + lcol;

    auto compute_tile = [&](int buf) {
        FragBF af[2];
#pragma unroll
        for (int fm = 0; fm < 2; ++fm) {
            const unsigned short* abase =
                AsBuf[buf] + (wm * 32 + fm * 16 + lmod) * 32 + lhalf * 8;
            af[fm].lo = *(const U128*)(abase);
            af[fm].hi = *(const U128*)(abase + 16);
        }
#pragma unroll
        for (int fn = 0; fn < 4; ++fn) {
            FragBF bf_;
            const unsigned short* bbase =
                BsBuf[buf] + (wn * 64 + fn * 16 + lmod) * 32 + lhalf * 8;
            bf_.lo = *(const U128*)(bbase);
            bf_.hi = *(const U128*)(bbase + 16);
#pragma unroll
            for (int fm = 0; fm < 2; ++fm) {
                acc[fm][fn] = __builtin_amdgcn_wmma_f32_16x16x32_bf16(
                    false, frag_cast(af[fm]), false, frag_cast(bf_),
                    (short)0, acc[fm][fn], false, false);
            }
        }
    };

#if USE_ASYNC_LDS
    auto issue_tile = [&](int ks, int buf) {
        async_cp32(gA0 + ks * 32, AsBuf[buf] + lrow * 32 + lcol);
        async_cp32(gB0 + ks * 32, BsBuf[buf] + lrow * 32 + lcol);
    };
    issue_tile(0, 0);
    for (int ks = 0; ks < INV_KSTEPS - 1; ++ks) {
        issue_tile(ks + 1, (ks + 1) & 1);        // prefetch next tile
        __builtin_amdgcn_s_wait_asynccnt(4);     // oldest tile (4 ops) done
        __syncthreads();
        compute_tile(ks & 1);
        __syncthreads();                         // readers done before reuse
    }
    __builtin_amdgcn_s_wait_asynccnt(0);
    __syncthreads();
    compute_tile((INV_KSTEPS - 1) & 1);
    __syncthreads();
#else
    for (int ks = 0; ks < INV_KSTEPS; ++ks) {
        {   const U128* g = (const U128*)(gA0 + ks * 32);
            U128* s = (U128*)(AsBuf[0] + lrow * 32 + lcol);
            s[0] = g[0]; s[1] = g[1]; }
        {   const U128* g = (const U128*)(gB0 + ks * 32);
            U128* s = (U128*)(BsBuf[0] + lrow * 32 + lcol);
            s[0] = g[0]; s[1] = g[1]; }
        if (ks + 1 < INV_KSTEPS) {
            __builtin_prefetch(gB0 + (ks + 1) * 32, 0, 0);
        }
        __syncthreads();
        compute_tile(0);
        __syncthreads();
    }
#endif

    // Epilogue: two 64-column halves; stage grid in LDS, project over h.
    for (int h = 0; h < 2; ++h) {
        if (wn == h) {
#pragma unroll
            for (int fm = 0; fm < 2; ++fm)
#pragma unroll
                for (int fn = 0; fn < 4; ++fn)
#pragma unroll
                    for (int r = 0; r < 8; ++r) {
                        const int ch = wm * 32 + fm * 16 + r + lhalf * 8;
                        Cls[ch * 68 + fn * 16 + lmod] = acc[fm][fn][r];
                    }
        }
        __syncthreads();
#pragma unroll
        for (int u = 0; u < 2; ++u) {
            const int idx = t + u * 256;   // 0..511 = 8o x 64c
            const int o = idx >> 6, c = idx & 63;
            float s = out_b[o];
            for (int ch = 0; ch < HID; ++ch)
                s += Wout[o * HID + ch] * Cls[ch * 68 + c];
            out[(size_t)(b * OUTC + o) * NPIX + n0 + h * 64 + c] = s;
        }
        __syncthreads();
    }
}

// ============================================================
// host launcher
// ============================================================
extern "C" void kernel_launch(void* const* d_in, const int* in_sizes, int n_in,
                              void* d_out, int out_size, void* d_ws, size_t ws_size,
                              hipStream_t stream) {
    (void)in_sizes; (void)n_in; (void)out_size; (void)ws_size;
    const float* x       = (const float*)d_in[0];
    const float* in_w    = (const float*)d_in[1];
    const float* in_b    = (const float*)d_in[2];
    const float* wr      = (const float*)d_in[3];
    const float* wi      = (const float*)d_in[4];
    const float* sp_bias = (const float*)d_in[5];
    const float* lin_w   = (const float*)d_in[6];
    const float* lin_b   = (const float*)d_in[7];
    const float* out_w   = (const float*)d_in[8];
    const float* out_b   = (const float*)d_in[9];
    const float* Y       = (const float*)d_in[10];
    const float* qw      = (const float*)d_in[11];

    char* ws = (char*)d_ws;
    size_t off = 0;
    auto carve = [&](size_t bytes) -> void* {
        void* p = ws + off;
        off += (bytes + 255) & ~(size_t)255;
        return p;
    };
    unsigned short* Ytbf = (unsigned short*)carve((size_t)NPIX * KTOT * 2);  // 85.0 MB
    unsigned short* Ybf  = (unsigned short*)carve((size_t)MODES * NPIX * 2); //  1.8 MB
    unsigned short* hq   = (unsigned short*)carve((size_t)BH * NPIX * 2);    // 75.5 MB
    float*          cr   = (float*)carve((size_t)BH * KTOT * 4);             // 18.9 MB
    float*          ci   = (float*)carve((size_t)BH * MODES * 4);            //  0.4 MB
    unsigned short* crbf = (unsigned short*)carve((size_t)BH * KTOT * 2);    //  9.4 MB

    (void)hipMemsetAsync(cr, 0, (size_t)BH * KTOT * 4, stream);
    (void)hipMemsetAsync(ci, 0, (size_t)BH * MODES * 4, stream);

    k_yconv  <<<dim3(NPIX / 32, KTOT / 32), 256, 0, stream>>>(Y, Ybf, Ytbf);
    k_encode <<<dim3(BATCH * (NPIX / 128)), 256, 0, stream>>>(x, in_w, in_b, qw, hq);
    k_fwd_sht<<<dim3(BH / 64, NPIX / 768), 128, 0, stream>>>(hq, Ybf, cr);
    for (int l = 0; l < NLAYERS; ++l)
        k_layer<<<dim3(MODES), 128, 0, stream>>>(wr, wi, sp_bias, lin_w, lin_b, cr, ci, l);
    k_tail   <<<dim3(KTOT - MODES), 128, 0, stream>>>(sp_bias, lin_w, lin_b, cr);
    k_crbf   <<<dim3((BH * KTOT) / 256), 256, 0, stream>>>(cr, crbf);
    k_inv_proj<<<dim3(NPIX / 128, BATCH), 256, 0, stream>>>(crbf, Ytbf, out_w, out_b,
                                                            (float*)d_out);
}
